// MemoryLSTM_52037823758456
// MI455X (gfx1250) — matrix operand order repaired and checked by reading, not compiled, not run on hardware
//
#include <hip/hip_runtime.h>
#include <hip/hip_bf16.h>
#include <math.h>

typedef __bf16 bf16_t;
typedef __attribute__((ext_vector_type(16))) __bf16 v16bf;
typedef __attribute__((ext_vector_type(8)))  __bf16 v8bf;
typedef __attribute__((ext_vector_type(8)))  float  v8f;

constexpr int T = 512;
constexpr int B = 128;
constexpr int I = 512;
constexpr int H = 1024;
constexpr int G = 4096;   // 4*H gates
constexpr int O = 256;

// ---------------------------------------------------------------------------
// f32 -> bf16 conversion (weights, one-time)
// ---------------------------------------------------------------------------
__global__ void k_f32_to_bf16(const float* __restrict__ in,
                              bf16_t* __restrict__ out, int n) {
    int i = blockIdx.x * blockDim.x + threadIdx.x;
    if (i < n) out[i] = (bf16_t)in[i];
}

// ---------------------------------------------------------------------------
// bias[g] = b_ih[g] + b_hh[g]   (one-time)
// ---------------------------------------------------------------------------
__global__ void k_sum_bias(const float* __restrict__ b_ih,
                           const float* __restrict__ b_hh,
                           float* __restrict__ bias, int n) {
    int i = blockIdx.x * blockDim.x + threadIdx.x;
    if (i < n) bias[i] = b_ih[i] + b_hh[i];
}

// ---------------------------------------------------------------------------
// zero-init h (bf16) and c (f32)
// ---------------------------------------------------------------------------
__global__ void k_init_state(float* __restrict__ c, bf16_t* __restrict__ h, int n) {
    int i = blockIdx.x * blockDim.x + threadIdx.x;
    if (i < n) { c[i] = 0.0f; h[i] = (bf16_t)0.0f; }
}

// ---------------------------------------------------------------------------
// One-time input projection for ALL timesteps (fully parallel, off the
// serial path):  xg[t*B+b, g] = sum_i x[t,b,i]*w_ih[g,i] + bias[g]  (bf16)
// Rows M = T*B = 65536, cols N = G = 4096 -> 1M 16x16 tiles.
// Each wave: 2 M-tiles sharing one B-fragment. 4 waves/block over N.
// Grid: 2048 M-pairs x 64 N-groups = 131072 blocks x 128 threads.
// ---------------------------------------------------------------------------
__global__ void k_xgates(const float*  __restrict__ x,     // [T*B, I] f32
                         const bf16_t* __restrict__ wih,   // [G, I]  bf16
                         const float*  __restrict__ bias,  // [G]     f32
                         bf16_t* __restrict__ xg) {        // [T*B, G] bf16
    const int lane  = threadIdx.x & 31;
    const int wave  = threadIdx.x >> 5;
    const int mpair = blockIdx.x >> 6;                 // 0..2047
    const int ntile = (blockIdx.x & 63) * 4 + wave;    // 0..255
    const int r     = lane & 15;
    const int half  = lane >> 4;
    const int kbA   = half * 16;
    const int kbB   = half * 16;
    const int gcol  = ntile * 16 + r;
    const int row0  = mpair * 32 + r;
    const int row1  = row0 + 16;

    v8f acc0 = {};
    v8f acc1 = {};
    const bf16_t* wrow = wih + (size_t)gcol * I;
    const float*  xr0  = x + (size_t)row0 * I;
    const float*  xr1  = x + (size_t)row1 * I;
    for (int k0 = 0; k0 < I; k0 += 32) {
        v16bf bfrag = *(const v16bf*)(wrow + k0 + kbB);
        v16bf a0, a1;
#pragma unroll
        for (int e = 0; e < 8; ++e) {
            a0[e]     = (bf16_t)xr0[k0 + kbA + e];
            a0[e + 8] = (bf16_t)xr0[k0 + kbA + 8 + e];
            a1[e]     = (bf16_t)xr1[k0 + kbA + e];
            a1[e + 8] = (bf16_t)xr1[k0 + kbA + 8 + e];
        }
        acc0 = __builtin_amdgcn_wmma_f32_16x16x32_bf16(
                   false, a0, false, bfrag, (short)0, acc0, false, false);
        acc1 = __builtin_amdgcn_wmma_f32_16x16x32_bf16(
                   false, a1, false, bfrag, (short)0, acc1, false, false);
    }
    const int   n  = ntile * 16 + r;      // D layout: lane owns column n
    const float bv = bias[n];
#pragma unroll
    for (int rr = 0; rr < 8; ++rr) {
        int m0 = mpair * 32 + half * 8 + rr;
        int m1 = m0 + 16;
        xg[(size_t)m0 * G + n] = (bf16_t)(acc0[rr] + bv);
        xg[(size_t)m1 * G + n] = (bf16_t)(acc1[rr] + bv);
    }
}

// ---------------------------------------------------------------------------
// Fused LSTM recurrence step (one launch per timestep) — serial-path only:
//   gates = xg[t] + h_in @ w_hh.T        (bf16 WMMA, f32 acc)
//   i,f,g,o = nonlinearities(gates)       (LDS tile exchange)
//   c = f*c + i*g ; h_out = o*tanh(c)
// Block = 128 threads (4 waves); wave q owns gate-quadrant q. Each wave:
// 2 M-tiles sharing each B-fragment (halves weight traffic; 2 independent
// WMMAs per fragment fill the matrix pipe). Grid: 4 M-pairs x 64 j-tiles.
// ---------------------------------------------------------------------------
__global__ void k_lstm_step(const bf16_t* __restrict__ xg,    // [T*B, G] bf16
                            const bf16_t* __restrict__ whh,   // [G, H]  bf16
                            const bf16_t* __restrict__ h_in,  // [B, H]  bf16
                            float*  __restrict__ c,           // [B, H]  f32
                            bf16_t* __restrict__ h_out,       // [B, H]  bf16
                            int t) {
    __shared__ float lds[2][4][256];   // [mSub][gate][16x16 patch]

    const int lane = threadIdx.x & 31;
    const int wave = threadIdx.x >> 5;      // gate quadrant q = 0..3
    const int mp   = blockIdx.x & 3;        // M-pair: batch rows [mp*32, mp*32+32)
    const int jn   = blockIdx.x >> 2;       // j-tile: hidden cols [jn*16, jn*16+16)
    const int r    = lane & 15;
    const int half = lane >> 4;
    const int kbA  = half * 16;
    const int kbB  = half * 16;

    const int gcol  = wave * H + jn * 16 + r;   // weight row (gate index)
    const int brow0 = (mp * 2)     * 16 + r;
    const int brow1 = (mp * 2 + 1) * 16 + r;

    v8f acc0 = {};
    v8f acc1 = {};

    // ---- recurrent projection: K over H ----
    const bf16_t* whhrow = whh + (size_t)gcol * H;
    const bf16_t* hrow0  = h_in + (size_t)brow0 * H;
    const bf16_t* hrow1  = h_in + (size_t)brow1 * H;
    for (int k0 = 0; k0 < H; k0 += 32) {
        v16bf bfrag = *(const v16bf*)(whhrow + k0 + kbB);
        v8bf p0 = *(const v8bf*)(hrow0 + k0 + kbA);
        v8bf q0 = *(const v8bf*)(hrow0 + k0 + kbA + 8);
        v8bf p1 = *(const v8bf*)(hrow1 + k0 + kbA);
        v8bf q1 = *(const v8bf*)(hrow1 + k0 + kbA + 8);
        v16bf a0, a1;
#pragma unroll
        for (int e = 0; e < 8; ++e) {
            a0[e] = p0[e]; a0[e + 8] = q0[e];
            a1[e] = p1[e]; a1[e + 8] = q1[e];
        }
        acc0 = __builtin_amdgcn_wmma_f32_16x16x32_bf16(
                   false, a0, false, bfrag, (short)0, acc0, false, false);
        acc1 = __builtin_amdgcn_wmma_f32_16x16x32_bf16(
                   false, a1, false, bfrag, (short)0, acc1, false, false);
    }

    // ---- D layout: lane holds col n=r, rows m = half*8+rr -> LDS patch ----
#pragma unroll
    for (int rr = 0; rr < 8; ++rr) {
        lds[0][wave][(half * 8 + rr) * 16 + r] = acc0[rr];
        lds[1][wave][(half * 8 + rr) * 16 + r] = acc1[rr];
    }
    __syncthreads();

    // ---- pointwise cell update; xg (with bias folded in) added here ----
    const bf16_t* xg_t = xg + (size_t)t * B * G;
#pragma unroll
    for (int e = threadIdx.x; e < 512; e += 128) {
        int ms  = e >> 8;                   // which M-tile of the pair
        int p   = e & 255;                  // element within 16x16 patch
        int row = p >> 4;
        int col = p & 15;
        int b   = (mp * 2 + ms) * 16 + row; // batch index
        int j   = jn * 16 + col;            // hidden index
        const bf16_t* xgb = xg_t + (size_t)b * G + j;
        float ig = lds[ms][0][p] + (float)xgb[0        ];
        float fg = lds[ms][1][p] + (float)xgb[H        ];
        float gg = lds[ms][2][p] + (float)xgb[2 * H    ];
        float og = lds[ms][3][p] + (float)xgb[3 * H    ];
        ig = 1.0f / (1.0f + __expf(-ig));
        fg = 1.0f / (1.0f + __expf(-fg));
        gg = tanhf(gg);
        og = 1.0f / (1.0f + __expf(-og));
        size_t idx = (size_t)b * H + j;
        float cn = fg * c[idx] + ig * gg;
        c[idx]     = cn;
        h_out[idx] = (bf16_t)(og * tanhf(cn));
    }
}

// ---------------------------------------------------------------------------
// Dense head: logits[b,o] = sum_k h[b,k]*w_dense[o,k] + b_dense[o]
// 8 M-tiles x 16 N-tiles = 128 tiles; one per wave; 16 blocks x 8 waves.
// ---------------------------------------------------------------------------
__global__ void k_dense_gemm(const bf16_t* __restrict__ h,   // [B,H]
                             const bf16_t* __restrict__ wd,  // [O,H]
                             const float*  __restrict__ bd,  // [O]
                             float* __restrict__ logits) {   // [B,O]
    const int lane  = threadIdx.x & 31;
    const int wave  = threadIdx.x >> 5;
    const int tile  = blockIdx.x * 8 + wave;
    const int tileM = tile & 7;
    const int tileN = tile >> 3;
    const int r     = lane & 15;
    const int half  = lane >> 4;
    const int brow  = tileM * 16 + r;
    const int ocol  = tileN * 16 + r;
    const int kbA   = half * 16;
    const int kbB   = half * 16;

    v8f acc = {};
    const bf16_t* hrow = h  + (size_t)brow * H;
    const bf16_t* wrow = wd + (size_t)ocol * H;
    for (int k0 = 0; k0 < H; k0 += 32) {
        v8bf a0 = *(const v8bf*)(hrow + k0 + kbA);
        v8bf a1 = *(const v8bf*)(hrow + k0 + kbA + 8);
        v16bf a;
#pragma unroll
        for (int e = 0; e < 8; ++e) { a[e] = a0[e]; a[e + 8] = a1[e]; }
        v16bf b = *(const v16bf*)(wrow + k0 + kbB);
        acc = __builtin_amdgcn_wmma_f32_16x16x32_bf16(
                  false, a, false, b, (short)0, acc, false, false);
    }
    const int n = tileN * 16 + r;
    const float biasv = bd[n];
#pragma unroll
    for (int rr = 0; rr < 8; ++rr) {
        int m = tileM * 16 + half * 8 + rr;
        logits[(size_t)m * O + n] = acc[rr] + biasv;
    }
}

// ---------------------------------------------------------------------------
// log_softmax over axis 1 (O=256). One block (256 threads) per row.
// ---------------------------------------------------------------------------
__global__ void k_log_softmax(const float* __restrict__ logits,
                              float* __restrict__ out) {
    __shared__ float red[256];
    const int row = blockIdx.x;
    const int tid = threadIdx.x;
    float v = logits[row * O + tid];
    red[tid] = v;
    __syncthreads();
    for (int s = 128; s > 0; s >>= 1) {
        if (tid < s) red[tid] = fmaxf(red[tid], red[tid + s]);
        __syncthreads();
    }
    float m = red[0];
    __syncthreads();
    red[tid] = __expf(v - m);
    __syncthreads();
    for (int s = 128; s > 0; s >>= 1) {
        if (tid < s) red[tid] += red[tid + s];
        __syncthreads();
    }
    float lse = m + logf(red[0]);
    out[row * O + tid] = v - lse;
}

// ---------------------------------------------------------------------------
extern "C" void kernel_launch(void* const* d_in, const int* in_sizes, int n_in,
                              void* d_out, int out_size, void* d_ws, size_t ws_size,
                              hipStream_t stream) {
    (void)in_sizes; (void)n_in; (void)out_size; (void)ws_size;
    const float* x       = (const float*)d_in[0];  // [T,B,I]
    const float* w_ih    = (const float*)d_in[1];  // [G,I]
    const float* w_hh    = (const float*)d_in[2];  // [G,H]
    const float* b_ih    = (const float*)d_in[3];  // [G]
    const float* b_hh    = (const float*)d_in[4];  // [G]
    const float* w_dense = (const float*)d_in[5];  // [O,H]
    const float* b_dense = (const float*)d_in[6];  // [O]
    float* out = (float*)d_out;                    // [B,O]

    char* ws = (char*)d_ws;
    size_t off = 0;
    auto carve = [&](size_t bytes) -> void* {
        void* p = ws + off;
        off = (off + bytes + 255) & ~(size_t)255;
        return p;
    };
    bf16_t* xg     = (bf16_t*)carve((size_t)T * B * G * sizeof(bf16_t)); // 512 MB
    bf16_t* wih_b  = (bf16_t*)carve((size_t)G * I * sizeof(bf16_t));     // 4 MB
    bf16_t* whh_b  = (bf16_t*)carve((size_t)G * H * sizeof(bf16_t));     // 8 MB
    bf16_t* wd_b   = (bf16_t*)carve((size_t)O * H * sizeof(bf16_t));     // 0.5 MB
    float*  bias   = (float*) carve((size_t)G * sizeof(float));          // 16 KB
    bf16_t* hbuf0  = (bf16_t*)carve((size_t)B * H * sizeof(bf16_t));     // 256 KB
    bf16_t* hbuf1  = (bf16_t*)carve((size_t)B * H * sizeof(bf16_t));     // 256 KB
    float*  c_st   = (float*) carve((size_t)B * H * sizeof(float));      // 512 KB
    float*  logits = (float*) carve((size_t)B * O * sizeof(float));      // 128 KB

    bf16_t* hbuf[2] = { hbuf0, hbuf1 };

    // one-time prep
    k_f32_to_bf16<<<(G * I + 255) / 256, 256, 0, stream>>>(w_ih, wih_b, G * I);
    k_f32_to_bf16<<<(G * H + 255) / 256, 256, 0, stream>>>(w_hh, whh_b, G * H);
    k_f32_to_bf16<<<(O * H + 255) / 256, 256, 0, stream>>>(w_dense, wd_b, O * H);
    k_sum_bias   <<<(G + 255) / 256, 256, 0, stream>>>(b_ih, b_hh, bias, G);
    k_init_state <<<(B * H + 255) / 256, 256, 0, stream>>>(c_st, hbuf[0], B * H);

    // one-time fully-parallel input projection for all T (bias folded in)
    k_xgates<<<(T * B / 32) * 64, 128, 0, stream>>>(x, wih_b, bias, xg);

    // sequential recurrence: one fused kernel per timestep, h double-buffered
    for (int t = 0; t < T; ++t) {
        k_lstm_step<<<256, 128, 0, stream>>>(xg, whh_b,
                                             hbuf[t & 1],
                                             c_st, hbuf[(t + 1) & 1], t);
    }

    // T even -> final hidden state is in hbuf[0]
    k_dense_gemm <<<16, 256, 0, stream>>>(hbuf[0], wd_b, b_dense, logits);
    k_log_softmax<<<B, 256, 0, stream>>>(logits, out);
}
